// Encoder_32212254720645
// MI455X (gfx1250) — matrix operand order repaired and checked by reading, not compiled, not run on hardware
//
#include <hip/hip_runtime.h>

typedef __attribute__((ext_vector_type(16))) _Float16 v16h;
typedef __attribute__((ext_vector_type(8)))  float    v8f;

#define WAVES_PER_BLOCK 8

// ---------------------------------------------------------------------------
// f32 -> f16 conversion (for the activation matrix A)
// ---------------------------------------------------------------------------
__global__ void cvt_f32_f16(const float* __restrict__ src, _Float16* __restrict__ dst, int count) {
    int i = blockIdx.x * blockDim.x + threadIdx.x;
    if (i < count) dst[i] = (_Float16)src[i];
}

// ---------------------------------------------------------------------------
// Pack weight matrix W[K x ncols] (f32) into WMMA B-fragment order (f16):
// Bp[((nt*(K/32) + kkt)*32 + lane)*16 + j] = W[(kkt*32 + 16*(lane>>4) + j)*ncols
//                                              + nt*16 + (lane&15)]
// -> in the GEMM each lane reads its 16 B elements as one contiguous 32B v16h.
// ---------------------------------------------------------------------------
__global__ void pack_B(const float* __restrict__ W, _Float16* __restrict__ Bp,
                       int K, int ncols) {
    int t = blockIdx.x * blockDim.x + threadIdx.x;
    const int kkt_count = K >> 5;
    const int tiles_n   = ncols >> 4;
    if (t >= tiles_n * kkt_count * 32) return;
    const int l   = t & 31;
    const int kkt = (t >> 5) % kkt_count;
    const int nt  = (t >> 5) / kkt_count;
    const int col  = nt * 16 + (l & 15);
    const int krow = kkt * 32 + 16 * (l >> 4);
    _Float16* dst = Bp + (size_t)t * 16;
#pragma unroll
    for (int j = 0; j < 16; ++j)
        dst[j] = (_Float16)W[(size_t)(krow + j) * ncols + col];
}

// ---------------------------------------------------------------------------
// degree + per-dst edge_attr sums (for self-loop fill_value='mean')
// ---------------------------------------------------------------------------
__global__ void deg_loop_kernel(const int* __restrict__ dst, const float* __restrict__ eattr,
                                int* __restrict__ deg, float* __restrict__ lsum, int E) {
    int e = blockIdx.x * blockDim.x + threadIdx.x;
    if (e >= E) return;
    int d = dst[e];
    atomicAdd(&deg[d], 1);
    atomicAdd(&lsum[d * 3 + 0], eattr[e * 3 + 0]);
    atomicAdd(&lsum[d * 3 + 1], eattr[e * 3 + 1]);
    atomicAdd(&lsum[d * 3 + 2], eattr[e * 3 + 2]);
}

// ---------------------------------------------------------------------------
// single-block exclusive scan of (deg+1) -> CSR offsets; normalize loop attrs
// ---------------------------------------------------------------------------
__global__ __launch_bounds__(1024)
void scan_kernel(const int* __restrict__ deg, float* __restrict__ lsum,
                 int* __restrict__ off, int n) {
    __shared__ int ps[1024];
    int t = threadIdx.x;
    int chunk = (n + 1023) >> 10;
    int s = t * chunk;
    int e = min(n, s + chunk);
    int sum = 0;
    for (int i = s; i < e; ++i) sum += deg[i] + 1;
    ps[t] = sum;
    __syncthreads();
    for (int d = 1; d < 1024; d <<= 1) {
        int v = (t >= d) ? ps[t - d] : 0;
        __syncthreads();
        ps[t] += v;
        __syncthreads();
    }
    int run = (t == 0) ? 0 : ps[t - 1];
    for (int i = s; i < e; ++i) {
        off[i] = run;
        run += deg[i] + 1;
        float invd = 1.0f / fmaxf((float)deg[i], 1.0f);
        lsum[i * 3 + 0] *= invd;
        lsum[i * 3 + 1] *= invd;
        lsum[i * 3 + 2] *= invd;
    }
    if (t == 1023) off[n] = ps[1023];
}

// ---------------------------------------------------------------------------
// CSR fill: real edges via per-dst cursor, self-loop id E+n in last slot
// ---------------------------------------------------------------------------
__global__ void csr_fill_kernel(const int* __restrict__ dst, const int* __restrict__ deg,
                                const int* __restrict__ off, int* __restrict__ cursor,
                                int* __restrict__ csr, int E, int N) {
    int t = blockIdx.x * blockDim.x + threadIdx.x;
    if (t < E) {
        int d = dst[t];
        int p = atomicAdd(&cursor[d], 1);
        csr[off[d] + p] = t;
    } else if (t < E + N) {
        int n = t - E;
        csr[off[n] + deg[n]] = E + n;   // self-loop edge id
    }
}

// ---------------------------------------------------------------------------
// WMMA GEMM: C[M x ncols] = A[M x K](f16) * Bpacked(f16) + bias
// one wave per 16x16 tile; K templated so the K-loop fully unrolls.
// Per K-step: 2x b128 A loads, 2x b128 B loads (packed), 1 wmma.
// Requires M%16==0, ncols%16==0 (M=50000, ncols in {256,32}).
// ---------------------------------------------------------------------------
template <int K>
__global__ __launch_bounds__(256)
void gemm_wmma_bias(const _Float16* __restrict__ A, const _Float16* __restrict__ Bp,
                    const float* __restrict__ bias, float* __restrict__ C,
                    int M, int ncols) {
    const int wid     = blockIdx.x * WAVES_PER_BLOCK + (threadIdx.x >> 5);
    const int tiles_n = ncols >> 4;
    if (wid >= (M >> 4) * tiles_n) return;           // wave-uniform: EXEC stays all-ones
    const int mt = wid / tiles_n, nt = wid % tiles_n;
    const int l = threadIdx.x & 31;
    const int r = l & 15;
    const int g = l >> 4;
    const int m0 = mt << 4, n0 = nt << 4;
    const int cb = n0 + r;

    const _Float16* pa = A + (size_t)(m0 + r) * K + 16 * g;
    const _Float16* pb = Bp + ((size_t)nt * (K >> 5) * 32 + l) * 16;

    v8f c = {0.f, 0.f, 0.f, 0.f, 0.f, 0.f, 0.f, 0.f};
#pragma unroll
    for (int kk = 0; kk < K; kk += 32) {
        v16h a;
#pragma unroll
        for (int j = 0; j < 8; ++j) {
            a[j]     = pa[j];        // K = kk + 16g + j
            a[8 + j] = pa[8 + j];    // K = kk + 16g + 8 + j
        }
        v16h b = *(const v16h*)pb;   // packed: 32 contiguous bytes per lane
        c = __builtin_amdgcn_wmma_f32_16x16x32_f16(false, a, false, b,
                                                   (short)0, c, false, false);
        pa += 32;
        pb += 512;                   // 32 lanes * 16 halves per K-step
    }

    const float bv = bias[cb];
    float* pC = C + (size_t)(m0 + 8 * g) * ncols + cb;   // VGPR r8 -> row m0 + 8g + r8
#pragma unroll
    for (int r8 = 0; r8 < 8; ++r8)
        pC[(size_t)r8 * ncols] = c[r8] + bv;
}

// ---------------------------------------------------------------------------
// per-edge attention logits: one wave per (edge | self-loop)
// alpha[e][h] = sum_c lrelu(xl[s]+xr[d]+ea@We)[h,c] * att[h,c]
// ---------------------------------------------------------------------------
__global__ __launch_bounds__(256)
void alpha_kernel(const int* __restrict__ src, const int* __restrict__ dst,
                  const float* __restrict__ eattr, const float* __restrict__ loop_attr,
                  const float* __restrict__ We, const float* __restrict__ att,
                  const float* __restrict__ xl, const float* __restrict__ xr,
                  float* __restrict__ alpha, int E, int N) {
    int wid = blockIdx.x * WAVES_PER_BLOCK + (threadIdx.x >> 5);
    if (wid >= E + N) return;
    int l = threadIdx.x & 31;
    int s, d;
    float ea0, ea1, ea2;
    if (wid < E) {
        s = src[wid]; d = dst[wid];
        ea0 = eattr[wid * 3]; ea1 = eattr[wid * 3 + 1]; ea2 = eattr[wid * 3 + 2];
    } else {
        int n = wid - E;
        s = n; d = n;
        ea0 = loop_attr[n * 3]; ea1 = loop_attr[n * 3 + 1]; ea2 = loop_attr[n * 3 + 2];
    }
    const int hc0 = l * 8;            // lane owns 8 contiguous of the 256 = H*C dims
    const int h   = l >> 3;
    const int c0  = (l & 7) * 8;
    const float* pxl = xl + (size_t)s * 256 + hc0;
    const float* pxr = xr + (size_t)d * 256 + hc0;
    float partial = 0.f;
#pragma unroll
    for (int j = 0; j < 8; ++j) {
        int hc  = hc0 + j;
        float ee = ea0 * We[hc] + ea1 * We[256 + hc] + ea2 * We[512 + hc];
        float v  = pxl[j] + pxr[j] + ee;
        v = (v > 0.f) ? v : 0.2f * v;
        partial += v * att[h * 64 + c0 + j];
    }
    partial += __shfl_xor(partial, 1, 32);
    partial += __shfl_xor(partial, 2, 32);
    partial += __shfl_xor(partial, 4, 32);
    if ((l & 7) == 0) alpha[(size_t)wid * 4 + h] = partial;
}

// ---------------------------------------------------------------------------
// per-node softmax + weighted gather aggregation: one wave per node.
// out[n,c] = relu( mean_h( sum_e softmax(alpha)[e,h] * xl[src(e),h,c] ) + bias[c] )
// ---------------------------------------------------------------------------
__global__ __launch_bounds__(256)
void aggregate_kernel(const int* __restrict__ src, const int* __restrict__ off,
                      const int* __restrict__ csr, const float* __restrict__ alpha,
                      const float* __restrict__ xl, const float* __restrict__ bias,
                      float* __restrict__ hout, _Float16* __restrict__ hhout,
                      int E, int N) {
    int n = blockIdx.x * WAVES_PER_BLOCK + (threadIdx.x >> 5);
    if (n >= N) return;
    int l = threadIdx.x & 31;
    int beg = off[n], cnt = off[n + 1] - beg;

    // pass 1: per-head max (lanes stride edges)
    float mx0 = -1e30f, mx1 = -1e30f, mx2 = -1e30f, mx3 = -1e30f;
    for (int j = l; j < cnt; j += 32) {
        const float* pa = alpha + (size_t)csr[beg + j] * 4;
        mx0 = fmaxf(mx0, pa[0]); mx1 = fmaxf(mx1, pa[1]);
        mx2 = fmaxf(mx2, pa[2]); mx3 = fmaxf(mx3, pa[3]);
    }
    for (int m = 16; m >= 1; m >>= 1) {
        mx0 = fmaxf(mx0, __shfl_xor(mx0, m, 32));
        mx1 = fmaxf(mx1, __shfl_xor(mx1, m, 32));
        mx2 = fmaxf(mx2, __shfl_xor(mx2, m, 32));
        mx3 = fmaxf(mx3, __shfl_xor(mx3, m, 32));
    }
    // pass 2: per-head sum of exp
    float s0 = 0.f, s1 = 0.f, s2 = 0.f, s3 = 0.f;
    for (int j = l; j < cnt; j += 32) {
        const float* pa = alpha + (size_t)csr[beg + j] * 4;
        s0 += __expf(pa[0] - mx0); s1 += __expf(pa[1] - mx1);
        s2 += __expf(pa[2] - mx2); s3 += __expf(pa[3] - mx3);
    }
    for (int m = 16; m >= 1; m >>= 1) {
        s0 += __shfl_xor(s0, m, 32); s1 += __shfl_xor(s1, m, 32);
        s2 += __shfl_xor(s2, m, 32); s3 += __shfl_xor(s3, m, 32);
    }
    const int h  = l >> 3;
    float mh  = (h == 0) ? mx0 : (h == 1) ? mx1 : (h == 2) ? mx2 : mx3;
    float ivh = 1.f / ((h == 0) ? s0 : (h == 1) ? s1 : (h == 2) ? s2 : s3);

    // pass 3: weighted gather of xl[src] (lane owns 8 contiguous hc dims)
    const int hc0 = l * 8;
    float acc[8] = {0.f, 0.f, 0.f, 0.f, 0.f, 0.f, 0.f, 0.f};
    for (int j = 0; j < cnt; ++j) {
        int e  = csr[beg + j];
        int sn = (e < E) ? src[e] : (e - E);
        float a = __expf(alpha[(size_t)e * 4 + h] - mh) * ivh;
        const float* px = xl + (size_t)sn * 256 + hc0;
#pragma unroll
        for (int q = 0; q < 8; ++q) acc[q] += a * px[q];
    }
    // head mean: sum over lanes differing in bits 3,4 (heads 0..3 of same c group)
#pragma unroll
    for (int q = 0; q < 8; ++q) {
        acc[q] += __shfl_xor(acc[q], 8, 32);
        acc[q] += __shfl_xor(acc[q], 16, 32);
    }
    if (l < 8) {
        int c0 = l * 8;
#pragma unroll
        for (int q = 0; q < 8; ++q) {
            float v = fmaxf(acc[q] * 0.25f + bias[c0 + q], 0.f);
            hout[(size_t)n * 64 + c0 + q]  = v;
            hhout[(size_t)n * 64 + c0 + q] = (_Float16)v;
        }
    }
}

// ---------------------------------------------------------------------------
extern "C" void kernel_launch(void* const* d_in, const int* in_sizes, int n_in,
                              void* d_out, int out_size, void* d_ws, size_t ws_size,
                              hipStream_t stream) {
    (void)n_in; (void)out_size; (void)ws_size;
    const float* x     = (const float*)d_in[0];
    const int*   eidx  = (const int*)d_in[1];
    const float* eattr = (const float*)d_in[2];
    const float* Wl[2]   = {(const float*)d_in[3],  (const float*)d_in[10]};
    const float* bl[2]   = {(const float*)d_in[4],  (const float*)d_in[11]};
    const float* Wr[2]   = {(const float*)d_in[5],  (const float*)d_in[12]};
    const float* br[2]   = {(const float*)d_in[6],  (const float*)d_in[13]};
    const float* We[2]   = {(const float*)d_in[7],  (const float*)d_in[14]};
    const float* att[2]  = {(const float*)d_in[8],  (const float*)d_in[15]};
    const float* bias[2] = {(const float*)d_in[9],  (const float*)d_in[16]};
    const float* Wmu = (const float*)d_in[17];
    const float* bmu = (const float*)d_in[18];

    const int IN = 256, HC = 256, C = 64, LAT = 32;
    const int N = in_sizes[0] / IN;
    const int E = in_sizes[1] / 2;
    const int Etot = E + N;
    const int* srcArr = eidx;
    const int* dstArr = eidx + E;

    // workspace carve-out (256B aligned slices)
    char*  ws  = (char*)d_ws;
    size_t cur = 0;
    auto alloc = [&](size_t bytes) -> char* {
        char* p = ws + cur;
        cur = (cur + bytes + 255) & ~(size_t)255;
        return p;
    };
    _Float16* xh     = (_Float16*)alloc((size_t)N * IN * 2);
    _Float16* wlh    = (_Float16*)alloc((size_t)IN * HC * 2);
    _Float16* wrh    = (_Float16*)alloc((size_t)IN * HC * 2);
    _Float16* wmuh   = (_Float16*)alloc((size_t)C * LAT * 2);
    float*    xlBuf  = (float*)alloc((size_t)N * HC * 4);
    float*    xrBuf  = (float*)alloc((size_t)N * HC * 4);
    float*    alphaB = (float*)alloc((size_t)Etot * 4 * 4);
    int*      deg    = (int*)alloc((size_t)N * 4);
    float*    lsum   = (float*)alloc((size_t)N * 3 * 4);   // becomes loop_attr after scan
    int*      offs   = (int*)alloc((size_t)(N + 1) * 4);
    int*      cursor = (int*)alloc((size_t)N * 4);
    int*      csr    = (int*)alloc((size_t)Etot * 4);
    float*    hBuf   = (float*)alloc((size_t)N * C * 4);
    _Float16* hhBuf  = (_Float16*)alloc((size_t)N * C * 2);

    hipMemsetAsync(deg, 0, (size_t)N * 4, stream);
    hipMemsetAsync(lsum, 0, (size_t)N * 12, stream);
    hipMemsetAsync(cursor, 0, (size_t)N * 4, stream);

    // graph structure (shared by both layers)
    deg_loop_kernel<<<(E + 255) / 256, 256, 0, stream>>>(dstArr, eattr, deg, lsum, E);
    scan_kernel<<<1, 1024, 0, stream>>>(deg, lsum, offs, N);
    csr_fill_kernel<<<(Etot + 255) / 256, 256, 0, stream>>>(dstArr, deg, offs, cursor, csr, E, N);

    // precision staging
    cvt_f32_f16<<<((size_t)N * IN + 255) / 256, 256, 0, stream>>>(x, xh, N * IN);
    {   // pack final-projection weights once
        int tp = (LAT / 16) * (C / 32) * 32;
        pack_B<<<(tp + 255) / 256, 256, 0, stream>>>(Wmu, wmuh, C, LAT);
    }

    for (int L = 0; L < 2; ++L) {
        const int K = (L == 0) ? IN : C;
        const _Float16* Ain = (L == 0) ? xh : hhBuf;
        const int tp = (HC / 16) * (K / 32) * 32;
        pack_B<<<(tp + 255) / 256, 256, 0, stream>>>(Wl[L], wlh, K, HC);
        pack_B<<<(tp + 255) / 256, 256, 0, stream>>>(Wr[L], wrh, K, HC);

        const int tiles = (N / 16) * (HC / 16);
        const int grid  = (tiles + WAVES_PER_BLOCK - 1) / WAVES_PER_BLOCK;
        if (L == 0) {
            gemm_wmma_bias<256><<<grid, 256, 0, stream>>>(Ain, wlh, bl[L], xlBuf, N, HC);
            gemm_wmma_bias<256><<<grid, 256, 0, stream>>>(Ain, wrh, br[L], xrBuf, N, HC);
        } else {
            gemm_wmma_bias<64><<<grid, 256, 0, stream>>>(Ain, wlh, bl[L], xlBuf, N, HC);
            gemm_wmma_bias<64><<<grid, 256, 0, stream>>>(Ain, wrh, br[L], xrBuf, N, HC);
        }

        alpha_kernel<<<(Etot + WAVES_PER_BLOCK - 1) / WAVES_PER_BLOCK, 256, 0, stream>>>(
            srcArr, dstArr, eattr, lsum, We[L], att[L], xlBuf, xrBuf, alphaB, E, N);
        aggregate_kernel<<<(N + WAVES_PER_BLOCK - 1) / WAVES_PER_BLOCK, 256, 0, stream>>>(
            srcArr, offs, csr, alphaB, xlBuf, bias[L], hBuf, hhBuf, E, N);
    }

    // final projection: out = h @ Wmu + bmu
    const int tilesF = (N / 16) * (LAT / 16);
    gemm_wmma_bias<64><<<(tilesF + WAVES_PER_BLOCK - 1) / WAVES_PER_BLOCK, 256, 0, stream>>>(
        hhBuf, wmuh, bmu, (float*)d_out, N, LAT);
}
